// DensityGridNN_14499809591758
// MI455X (gfx1250) — matrix operand order repaired and checked by reading, not compiled, not run on hardware
//
#include <hip/hip_runtime.h>
#include <hip/hip_bf16.h>

typedef __attribute__((ext_vector_type(16))) _Float16 v16h;
typedef __attribute__((ext_vector_type(8)))  float    v8f;

// ---------------------------------------------------------------------------
// Stage feature projection, fused masking + transpose + f16 pack:
//   featsT[b, c16, n] = (c16 < C && n < lengths[b])
//                       ? f16(node_features[b,n,:] @ proj_w[:,c16] + proj_b[c16])
//                       : 0
// Layout (B, 16, N) f16, zero-padded so the WMMA B-fragment needs no masking
// and each 32-atom chunk is one contiguous 32-byte load per lane.
// ---------------------------------------------------------------------------
__global__ void proj_feats_kernel(const float* __restrict__ nf,
                                  const float* __restrict__ pw,
                                  const float* __restrict__ pb,
                                  const int*  __restrict__ lengths,
                                  _Float16* __restrict__ featsT,
                                  int B, int N, int E, int C)
{
    int idx = blockIdx.x * blockDim.x + threadIdx.x;   // over B*16*N
    int total = B * 16 * N;
    if (idx >= total) return;
    int n = idx % N;
    int c = (idx / N) % 16;
    int b = idx / (16 * N);
    float acc = 0.0f;
    if (c < C && n < lengths[b]) {
        acc = pb[c];
        const float* row = nf + (long)(b * N + n) * E;
        for (int e = 0; e < E; ++e) acc += row[e] * pw[e * C + c];
    }
    featsT[idx] = (_Float16)acc;
}

// ---------------------------------------------------------------------------
// Lorentzian scatter einsum on the matrix pipe (nothing else fused):
//   xg[b,c,v] = sum_n  sc/(sc^2 + |g_v - pos_n|^2) * mask_n * feats[b,n,c]
// One wave per 16-grid-point tile. A (16x32 f16) built in-register from
// positions via hardware v_rcp (branch-free; mask folded into featsT).
// B fragment = 32 contiguous bytes/lane from featsT. K=128 fully unrolled
// into 4 v_wmma_f32_16x16x32_f16. Live set ~40 VGPRs -> no spills.
// A/B use contiguous k = 16*half + e (not the ISA table order); D layout per CDNA5 ISA 7.12.2 (wave32).
// ---------------------------------------------------------------------------
template <int NATOMS>
__global__ void lorentz_scatter_wmma_kernel(
    const float* __restrict__ pos,       // (B,N,3)
    const float* __restrict__ origin,    // (B,3)
    const float* __restrict__ lattice,   // (B,3,3)
    const float* __restrict__ scale,     // (3,)
    const _Float16* __restrict__ featsT, // (B,16,N) masked+padded
    float* __restrict__ xout,            // (B,C,n,n,n)
    int B, int C, int n, float inv, int stage)
{
    const int lane = threadIdx.x & 31;
    const int wave = threadIdx.x >> 5;
    const int tilesPerBatch = (n * n * n) >> 4;
    const int tile = blockIdx.x * (blockDim.x >> 5) + wave;
    const int b = tile / tilesPerBatch;
    if (b >= B) return;                  // uniform per wave (exact division)
    const int t  = tile % tilesPerBatch;
    const long V = (long)n * n * n;
    const int v0 = t << 4;
    const int half = lane >> 4;          // which 16-lane half
    const int l16  = lane & 15;

    // grid point for this lane's A-matrix row (M = l16)
    const int vm = v0 + l16;
    const int iz = vm % n;
    const int iy = (vm / n) % n;
    const int ix = vm / (n * n);
    const float gx = origin[b * 3 + 0] + lattice[b * 9 + 0] * ((float)ix * inv);
    const float gy = origin[b * 3 + 1] + lattice[b * 9 + 4] * ((float)iy * inv);
    const float gz = origin[b * 3 + 2] + lattice[b * 9 + 8] * ((float)iz * inv);
    const float sc  = scale[stage];
    const float sc2 = sc * sc;

    // lane's B-fragment source row: channel column l16, half-selected K range
    const _Float16* brow = featsT + ((long)(b * 16 + l16)) * NATOMS + 16 * half;
    const float* prow = pos + (long)b * NATOMS * 3;

    v8f acc = {};
    #pragma unroll
    for (int kc = 0; kc < NATOMS / 32; ++kc) {
        const int a0 = kc * 32;
        union { v16h v; _Float16 h[16]; } Am, Bm;
        // A-matrix 16x32 f16: lanes<16 hold K {0..15}, lanes>=16 {16..31}
        #pragma unroll
        for (int e = 0; e < 16; ++e) {
            const int K = e + 16 * half;
            const float* pp = prow + (long)(a0 + K) * 3;
            const float dx = gx - pp[0];
            const float dy = gy - pp[1];
            const float dz = gz - pp[2];
            const float d2 = sc2 + dx * dx + dy * dy + dz * dz;
            Am.h[e] = (_Float16)(sc * __builtin_amdgcn_rcpf(d2));
        }
        // B-matrix 32x16 f16: 16 consecutive f16 (32 bytes) per lane
        Bm.v = *(const v16h*)(brow + a0);
        acc = __builtin_amdgcn_wmma_f32_16x16x32_f16(
            /*neg_a=*/false, Am.v, /*neg_b=*/false, Bm.v,
            /*c_mod=*/(short)0, acc, /*reuse_a=*/false, /*reuse_b=*/false);
    }

    // D layout: VGPR r -> (M = r + 8*half, channel = l16)
    if (l16 < C) {
        float* dst = xout + ((long)(b * C + l16)) * V + v0;
        #pragma unroll
        for (int r = 0; r < 8; ++r)
            dst[r + 8 * half] = acc[r];
    }
}

// ---------------------------------------------------------------------------
// Trilinear 2x upsample-add (half-pixel centers, edge clamp), in place:
//   x[b,c,ix,iy,iz] += trilerp(prev[b,c], (ix,iy,iz))
// ---------------------------------------------------------------------------
__global__ void upsample_add_kernel(const float* __restrict__ prev,
                                    float* __restrict__ x,
                                    int B, int C, int n)
{
    const long V = (long)n * n * n;
    const long total = (long)B * C * V;
    const long idx = (long)blockIdx.x * blockDim.x + threadIdx.x;
    if (idx >= total) return;
    int iz = (int)(idx % n);
    long tt = idx / n;
    int iy = (int)(tt % n); tt /= n;
    int ix = (int)(tt % n);
    int bc = (int)(tt / n);

    const int h = n >> 1;
    const long h3 = (long)h * h * h;
    const float* p = prev + (long)bc * h3;

    float sx = (ix + 0.5f) * 0.5f - 0.5f;
    float sy = (iy + 0.5f) * 0.5f - 0.5f;
    float sz = (iz + 0.5f) * 0.5f - 0.5f;
    int x0 = (int)floorf(sx); float fx = sx - (float)x0;
    int y0 = (int)floorf(sy); float fy = sy - (float)y0;
    int z0 = (int)floorf(sz); float fz = sz - (float)z0;
    int x0c = max(x0, 0), x1c = min(x0 + 1, h - 1);
    int y0c = max(y0, 0), y1c = min(y0 + 1, h - 1);
    int z0c = max(z0, 0), z1c = min(z0 + 1, h - 1);
    long hh = (long)h * h;
#define AT(X, Y, Z) p[(long)(X) * hh + (long)(Y) * h + (Z)]
    float c000 = AT(x0c, y0c, z0c), c001 = AT(x0c, y0c, z1c);
    float c010 = AT(x0c, y1c, z0c), c011 = AT(x0c, y1c, z1c);
    float c100 = AT(x1c, y0c, z0c), c101 = AT(x1c, y0c, z1c);
    float c110 = AT(x1c, y1c, z0c), c111 = AT(x1c, y1c, z1c);
#undef AT
    float c00 = c000 + (c001 - c000) * fz;
    float c01 = c010 + (c011 - c010) * fz;
    float c10 = c100 + (c101 - c100) * fz;
    float c11 = c110 + (c111 - c110) * fz;
    float c0 = c00 + (c01 - c00) * fy;
    float c1 = c10 + (c11 - c10) * fy;
    x[idx] += c0 + (c1 - c0) * fx;
}

// ---------------------------------------------------------------------------
// 3x3x3 circular conv, C->C, optional residual skip, ReLU output.
// mode 0: out = relu(conv(in) + bias)
// mode 1: out = relu(skip + conv(in) + bias)
// ---------------------------------------------------------------------------
__global__ void conv3_circ_kernel(const float* __restrict__ in,
                                  const float* __restrict__ w,
                                  const float* __restrict__ bias,
                                  const float* __restrict__ skip,
                                  float* __restrict__ out,
                                  int B, int C, int n, int mode)
{
    const long V = (long)n * n * n;
    const long total = (long)B * C * V;
    const long idx = (long)blockIdx.x * blockDim.x + threadIdx.x;
    if (idx >= total) return;
    int iz = (int)(idx % n);
    long tt = idx / n;
    int iy = (int)(tt % n); tt /= n;
    int ix = (int)(tt % n); tt /= n;
    int o  = (int)(tt % C);
    int b  = (int)(tt / C);

    const int xs[3] = { (ix == 0) ? n - 1 : ix - 1, ix, (ix == n - 1) ? 0 : ix + 1 };
    const int ys[3] = { (iy == 0) ? n - 1 : iy - 1, iy, (iy == n - 1) ? 0 : iy + 1 };
    const int zs[3] = { (iz == 0) ? n - 1 : iz - 1, iz, (iz == n - 1) ? 0 : iz + 1 };

    float accv = bias[o];
    const float* wo = w + (long)o * C * 27;
    for (int i = 0; i < C; ++i) {
        const float* inc = in + ((long)(b * C + i)) * V;
        const float* wi = wo + i * 27;
        int t27 = 0;
        #pragma unroll
        for (int kd = 0; kd < 3; ++kd) {
            const long xoff = (long)xs[kd] * n * n;
            #pragma unroll
            for (int kh = 0; kh < 3; ++kh) {
                const long xyoff = xoff + (long)ys[kh] * n;
                #pragma unroll
                for (int kw = 0; kw < 3; ++kw)
                    accv += wi[t27++] * inc[xyoff + zs[kw]];
            }
        }
    }
    if (mode == 1) accv += skip[idx];
    out[idx] = fmaxf(accv, 0.0f);
}

// ---------------------------------------------------------------------------
// 1x1x1 conv: Ci -> Co
// ---------------------------------------------------------------------------
__global__ void conv1x1_kernel(const float* __restrict__ in,
                               const float* __restrict__ w,
                               const float* __restrict__ bias,
                               float* __restrict__ out,
                               int B, int Ci, int Co, long V)
{
    const long total = (long)B * Co * V;
    const long idx = (long)blockIdx.x * blockDim.x + threadIdx.x;
    if (idx >= total) return;
    const long v = idx % V;
    const int co = (int)((idx / V) % Co);
    const int b  = (int)(idx / (V * Co));
    float accv = bias[co];
    for (int ci = 0; ci < Ci; ++ci)
        accv += w[co * Ci + ci] * in[((long)(b * Ci + ci)) * V + v];
    out[idx] = accv;
}

// ---------------------------------------------------------------------------
// Host launcher
// ---------------------------------------------------------------------------
extern "C" void kernel_launch(void* const* d_in, const int* in_sizes, int n_in,
                              void* d_out, int out_size, void* d_ws, size_t ws_size,
                              hipStream_t stream)
{
    (void)in_sizes; (void)n_in; (void)out_size; (void)ws_size;

    constexpr int B = 4, N = 128, E = 128;
    constexpr int CH[4] = {32, 16, 8, 1};

    const float* node_features = (const float*)d_in[0];
    const float* pos           = (const float*)d_in[1];
    const float* origin        = (const float*)d_in[2];
    const float* lattice       = (const float*)d_in[3];
    const float* scale         = (const float*)d_in[4];
    const float* proj_w[3] = {(const float*)d_in[5],  (const float*)d_in[6],  (const float*)d_in[7]};
    const float* proj_b[3] = {(const float*)d_in[8],  (const float*)d_in[9],  (const float*)d_in[10]};
    const float* res_w[3][2] = {{(const float*)d_in[11], (const float*)d_in[12]},
                                {(const float*)d_in[13], (const float*)d_in[14]},
                                {(const float*)d_in[15], (const float*)d_in[16]}};
    const float* res_b[3][2] = {{(const float*)d_in[17], (const float*)d_in[18]},
                                {(const float*)d_in[19], (const float*)d_in[20]},
                                {(const float*)d_in[21], (const float*)d_in[22]}};
    const float* out_w[3] = {(const float*)d_in[23], (const float*)d_in[24], (const float*)d_in[25]};
    const float* out_b[3] = {(const float*)d_in[26], (const float*)d_in[27], (const float*)d_in[28]};
    const int*   lengths  = (const int*)d_in[29];

    float* ws = (float*)d_ws;
    constexpr long BIG = 8388608;           // 4 * 8 * 64^3 floats (32 MB)
    _Float16* featsT = (_Float16*)ws;       // B*16*N f16 = 8192 halves (16 KB)
    float* W0 = ws + 16384;                 // keep 32B-aligned offsets
    float* W1 = W0 + BIG;
    float* W2 = W1 + BIG;

    // Buffer rotation (no reader/writer aliasing within a stage):
    //   s0: X=W1, H=W2, Y=W0, newP=W1
    //   s1: P=W1, X=W2, H=W0, Y=W1, newP=W0
    //   s2: P=W0, X=W1, H=W2, Y=W0, final -> d_out
    float* Pbuf[3]  = {nullptr, W1, W0};
    float* Xbuf[3]  = {W1, W2, W1};
    float* Hbuf[3]  = {W2, W0, W2};
    float* Ybuf[3]  = {W0, W1, W0};
    float* NPbuf[3] = {W1, W0, (float*)d_out};

    for (int s = 0; s < 3; ++s) {
        const int C  = CH[s];
        const int Cn = CH[s + 1];
        const int n  = 64 >> (2 - s);       // 16, 32, 64
        const long V = (long)n * n * n;
        const float inv = 1.0f / (float)(n - 1);

        // 1) feature projection (masked + transposed + f16-packed)
        {
            int total = B * 16 * N;
            proj_feats_kernel<<<(total + 255) / 256, 256, 0, stream>>>(
                node_features, proj_w[s], proj_b[s], lengths, featsT, B, N, E, C);
        }
        // 2) WMMA Lorentz scatter
        {
            int tiles = B * (int)(V >> 4);  // divisible by 8 for all stages
            int blocks = tiles / 8;
            lorentz_scatter_wmma_kernel<N><<<blocks, 256, 0, stream>>>(
                pos, origin, lattice, scale, featsT, Xbuf[s],
                B, C, n, inv, s);
        }
        // 2b) pyramid upsample-add (stages 1,2), in place on X
        if (s > 0) {
            long total = (long)B * C * V;
            int blocks = (int)((total + 255) / 256);
            upsample_add_kernel<<<blocks, 256, 0, stream>>>(
                Pbuf[s], Xbuf[s], B, C, n);
        }
        // 3) ResBlock: h = relu(conv(x)); y = relu(x + conv(h))
        {
            long total = (long)B * C * V;
            int blocks = (int)((total + 255) / 256);
            conv3_circ_kernel<<<blocks, 256, 0, stream>>>(
                Xbuf[s], res_w[s][0], res_b[s][0], nullptr, Hbuf[s], B, C, n, 0);
            conv3_circ_kernel<<<blocks, 256, 0, stream>>>(
                Hbuf[s], res_w[s][1], res_b[s][1], Xbuf[s], Ybuf[s], B, C, n, 1);
        }
        // 4) 1x1 conv -> next-stage input (or final output, Cn == 1)
        {
            long total = (long)B * Cn * V;
            int blocks = (int)((total + 255) / 256);
            conv1x1_kernel<<<blocks, 256, 0, stream>>>(
                Ybuf[s], out_w[s], out_b[s], NPbuf[s], B, C, Cn, V);
        }
    }
}
